// GraphTransformer_85830626443804
// MI455X (gfx1250) — matrix operand order repaired and checked by reading, not compiled, not run on hardware
//
#include <hip/hip_runtime.h>
#include <hip/hip_bf16.h>
#include <math.h>

typedef __attribute__((ext_vector_type(16))) _Float16 v16h;
typedef __attribute__((ext_vector_type(8)))  _Float16 v8h;
typedef __attribute__((ext_vector_type(8)))  float    v8f;

#define NN   25000
#define NNP  25024          // padded row count (multiple of 64); pad rows are zeroed
#define MTG  (NNP / 64)     // 391 row-groups of 64
#define NE   100000
#define NG   1024
#define XDIM 11
#define HIDD 128

// ---------- helpers ----------
static __device__ __forceinline__ void atomicMaxF(float* addr, float val) {
    if (val >= 0.f) atomicMax((int*)addr, __float_as_int(val));
    else            atomicMin((unsigned int*)addr, __float_as_uint(val));
}

__global__ void k_fill_f32(float* p, float v, int n) {
    int i = blockIdx.x * blockDim.x + threadIdx.x;
    if (i < n) p[i] = v;
}

// ---------- degree + input embedding ----------
__global__ void k_degree(const int* __restrict__ ei, int* __restrict__ deg) {
    int e = blockIdx.x * blockDim.x + threadIdx.x;
    if (e < NE) atomicAdd(&deg[ei[e]], 1); // src row
}

// h0[i] = concat(x[i][0:11], emb[clip(deg)][0:128]) zero-padded to K=160, f16; pad rows zero
__global__ void k_build_h0(const float* __restrict__ x, const float* __restrict__ emb,
                           const int* __restrict__ deg, _Float16* __restrict__ h, int lda) {
    int idx = blockIdx.x * blockDim.x + threadIdx.x;
    if (idx >= NNP * 160) return;
    int i = idx / 160, c = idx % 160;
    float v = 0.f;
    if (i < NN) {
        if (c < XDIM) v = x[i * XDIM + c];
        else if (c < XDIM + HIDD) {
            int d = deg[i]; if (d > 99) d = 99;
            v = emb[d * HIDD + (c - XDIM)];
        }
    }
    h[(size_t)i * lda + c] = (_Float16)v;
}

// ---------- weight prepack into per-block-contiguous WMMA B-panel layout ----------
// linear index = ((((ntg*Ktiles)+kt)*2 + ntl)*32 + lane)*16 + hh
// n = (ntg*2+ntl)*16 + (lane&15);  k = kt*32 + (lane<16 ? 0 : 8) + (hh<8 ? hh : 8+hh)
__global__ void k_prepack(const float* __restrict__ W, _Float16* __restrict__ pack,
                          int Kin, int Nout, int Ktiles, int elems) {
    int idx = blockIdx.x * blockDim.x + threadIdx.x;
    if (idx >= elems) return;
    int hh   = idx & 15;
    int lane = (idx >> 4) & 31;
    int rest = idx >> 9;
    int ntl = rest & 1;
    int t2  = rest >> 1;
    int kt  = t2 % Ktiles;
    int ntg = t2 / Ktiles;
    int n = (ntg * 2 + ntl) * 16 + (lane & 15);
    int k = kt * 32 + ((lane < 16) ? 0 : 8) + ((hh < 8) ? hh : 8 + hh);
    float v = (k < Kin) ? W[(size_t)k * Nout + n] : 0.f;
    pack[idx] = (_Float16)v;
}

// ---------- WMMA GEMM: C[NNP,N] = A_f16[NNP,Kpad] * Bpack + bias ----------
// block = 8 waves; each wave computes a 64x32 tile (4 M-tiles x 2 N-tiles, 8 accumulators).
// B panel for the block's 32 columns is async-DMA'd to LDS (ASYNCcnt), read back via ds_load.
__global__ __launch_bounds__(256) void k_gemm_wmma(const _Float16* __restrict__ A, int lda,
                                                   const _Float16* __restrict__ Bp, int Ktiles,
                                                   int N, int mtgGroups,
                                                   const float* __restrict__ bias,
                                                   float* __restrict__ C) {
    __shared__ _Float16 sB[16384]; // 32 KB: Ktiles(<=16) * 2 tiles * 32 lanes * 16 halves
    int ntg = blockIdx.x / mtgGroups;
    int mtg = (blockIdx.x % mtgGroups) * 8 + (threadIdx.x >> 5);
    int lane = threadIdx.x & 31;

    // ---- cooperative async stage of this block's B panel into LDS ----
    const char* panel = (const char*)(Bp + (size_t)ntg * Ktiles * 1024);
    int bytes = Ktiles * 2048;
    for (int o = threadIdx.x * 16; o < bytes; o += 256 * 16) {
        unsigned dst = (unsigned)(size_t)((char*)sB + o);
        unsigned long long src = (unsigned long long)(size_t)(panel + o);
        asm volatile("global_load_async_to_lds_b128 %0, %1, off"
                     :: "v"(dst), "v"(src) : "memory");
    }
    asm volatile("s_wait_asynccnt 0x0" ::: "memory");
    __syncthreads();

    if (mtg >= MTG) return; // wave-uniform

    const _Float16* arow0 = A + (size_t)(mtg * 64 + (lane & 15)) * lda + ((lane < 16) ? 0 : 8);

    v8f acc[2][4] = {};
    for (int kt = 0; kt < Ktiles; ++kt) {
        v16h b0 = *(const v16h*)(sB + ((size_t)(kt * 2 + 0) * 32 + lane) * 16);
        v16h b1 = *(const v16h*)(sB + ((size_t)(kt * 2 + 1) * 32 + lane) * 16);
#pragma unroll
        for (int mi = 0; mi < 4; ++mi) {
            const _Float16* ap = arow0 + (size_t)mi * 16 * lda + kt * 32;
            __builtin_prefetch(ap + 32, 0, 1);
            v8h lo = *(const v8h*)ap;        // K +0..7   (lanes>=16: +8..15)
            v8h hi = *(const v8h*)(ap + 16); // K +16..23 (lanes>=16: +24..31)
            v16h a;
#pragma unroll
            for (int i = 0; i < 8; ++i) { a[i] = lo[i]; a[i + 8] = hi[i]; }
            acc[0][mi] = __builtin_amdgcn_wmma_f32_16x16x32_f16(false, a, false, b0,
                                                                (short)0, acc[0][mi], false, false);
            acc[1][mi] = __builtin_amdgcn_wmma_f32_16x16x32_f16(false, a, false, b1,
                                                                (short)0, acc[1][mi], false, false);
        }
    }

    int col0 = ntg * 32 + (lane & 15);
    float bi0 = bias[col0], bi1 = bias[col0 + 16];
    int rbase = mtg * 64 + ((lane < 16) ? 0 : 8);
#pragma unroll
    for (int mi = 0; mi < 4; ++mi) {
#pragma unroll
        for (int r = 0; r < 8; ++r) {
            int rr = rbase + mi * 16 + r;
            C[(size_t)rr * N + col0]      = acc[0][mi][r] + bi0;
            C[(size_t)rr * N + col0 + 16] = acc[1][mi][r] + bi1;
        }
    }
}

// ---------- attention ----------
__global__ void k_edge_score(const float* __restrict__ q, const float* __restrict__ kk,
                             const int* __restrict__ ei, float* __restrict__ score,
                             float* __restrict__ mbuf, int H, int S) {
    int wid  = blockIdx.x * (blockDim.x >> 5) + (threadIdx.x >> 5);
    int lane = threadIdx.x & 31;
    if (wid >= NE * H) return;
    int e = wid / H, h = wid % H;
    int src = ei[e], dst = ei[NE + e];
    const float* qp = q  + (size_t)dst * S + h * HIDD;
    const float* kp = kk + (size_t)src * S + h * HIDD;
    float acc = 0.f;
#pragma unroll
    for (int d = lane; d < HIDD; d += 32) acc += qp[d] * kp[d];
    for (int m = 16; m; m >>= 1) acc += __shfl_xor(acc, m);
    if (lane == 0) {
        float sc = acc * 0.088388347648318447f; // 1/sqrt(128)
        score[wid] = sc;
        atomicMaxF(&mbuf[dst * H + h], sc);
    }
}

__global__ void k_exp_sum(float* __restrict__ score, const float* __restrict__ mbuf,
                          float* __restrict__ denom, const int* __restrict__ ei, int H) {
    int idx = blockIdx.x * blockDim.x + threadIdx.x;
    if (idx >= NE * H) return;
    int e = idx / H, h = idx % H;
    int dst = ei[NE + e];
    float v = __expf(score[idx] - mbuf[dst * H + h]);
    score[idx] = v;
    atomicAdd(&denom[dst * H + h], v);
}

__global__ void k_aggregate(const float* __restrict__ score, const float* __restrict__ denom,
                            const float* __restrict__ vv, const int* __restrict__ ei,
                            float* __restrict__ out, int H, int S) {
    int wid  = blockIdx.x * (blockDim.x >> 5) + (threadIdx.x >> 5);
    int lane = threadIdx.x & 31;
    if (wid >= NE * H) return;
    int e = wid / H, h = wid % H;
    int src = ei[e], dst = ei[NE + e];
    float alpha = score[wid] / denom[dst * H + h];
    const float* vp = vv + (size_t)src * S + h * HIDD;
    float* op = out + (size_t)dst * S + h * HIDD;
#pragma unroll
    for (int d = lane; d < HIDD; d += 32) atomicAdd(&op[d], alpha * vp[d]);
}

__global__ void k_elu_cvt(const float* __restrict__ src, _Float16* __restrict__ dst, int n) {
    int i = blockIdx.x * blockDim.x + threadIdx.x;
    if (i >= n) return;
    float x = src[i];
    float y = x > 0.f ? x : (__expf(x) - 1.f);
    dst[i] = (_Float16)y;
}

// ---------- pooling + MLP head ----------
__global__ void k_pool(const float* __restrict__ h3, const int* __restrict__ batch,
                       float* __restrict__ sums, float* __restrict__ cnts) {
    int idx = blockIdx.x * blockDim.x + threadIdx.x;
    if (idx >= NN * HIDD) return;
    int i = idx / HIDD, c = idx % HIDD;
    int b = batch[i];
    atomicAdd(&sums[b * HIDD + c], h3[(size_t)i * HIDD + c]);
    if (c == 0) atomicAdd(&cnts[b], 1.f);
}

__global__ void k_head(const float* __restrict__ sums, const float* __restrict__ cnts,
                       const float* __restrict__ w1, const float* __restrict__ b1,
                       const float* __restrict__ w2, const float* __restrict__ b2,
                       float* __restrict__ out) {
    int wid  = blockIdx.x * (blockDim.x >> 5) + (threadIdx.x >> 5);
    int lane = threadIdx.x & 31;
    if (wid >= NG) return;
    float inv = 1.f / fmaxf(cnts[wid], 1.f);
    float m0 = sums[wid * HIDD + lane]      * inv;
    float m1 = sums[wid * HIDD + lane + 32] * inv;
    float m2 = sums[wid * HIDD + lane + 64] * inv;
    float m3 = sums[wid * HIDD + lane + 96] * inv;
    float acc = 0.f;
    for (int j = 0; j < 64; ++j) {
        float p = m0 * w1[lane * 64 + j] + m1 * w1[(lane + 32) * 64 + j]
                + m2 * w1[(lane + 64) * 64 + j] + m3 * w1[(lane + 96) * 64 + j];
        for (int m = 16; m; m >>= 1) p += __shfl_xor(p, m);
        if (lane == 0) acc += fmaxf(p + b1[j], 0.f) * w2[j];
    }
    if (lane == 0) out[wid] = acc + b2[0];
}

// ---------- launch ----------
extern "C" void kernel_launch(void* const* d_in, const int* in_sizes, int n_in,
                              void* d_out, int out_size, void* d_ws, size_t ws_size,
                              hipStream_t stream) {
    (void)in_sizes; (void)n_in; (void)out_size; (void)ws_size;
    const float* x    = (const float*)d_in[0];
    const int*   ei   = (const int*)  d_in[1];
    const int*   batch= (const int*)  d_in[2];
    const float* emb  = (const float*)d_in[3];
    const float* W[12]; const float* B[12];
    for (int i = 0; i < 12; ++i) { W[i] = (const float*)d_in[4 + 2 * i]; B[i] = (const float*)d_in[5 + 2 * i]; }
    const float* fc1w = (const float*)d_in[28]; const float* fc1b = (const float*)d_in[29];
    const float* fc2w = (const float*)d_in[30]; const float* fc2b = (const float*)d_in[31];
    float* out = (float*)d_out;

    char* base = (char*)d_ws; size_t off = 0;
    auto nxt = [&](size_t bytes) -> void* { void* p = base + off; off += (bytes + 255) & ~(size_t)255; return p; };
    int*       deg   = (int*)      nxt((size_t)NN * 4);
    _Float16*  hf    = (_Float16*) nxt((size_t)NNP * 512 * 2);
    float*     qb    = (float*)    nxt((size_t)NNP * 512 * 4);
    float*     kb    = (float*)    nxt((size_t)NNP * 512 * 4);
    float*     vb    = (float*)    nxt((size_t)NNP * 512 * 4);
    float*     ob    = (float*)    nxt((size_t)NNP * 512 * 4);
    float*     score = (float*)    nxt((size_t)NE * 4 * 4);
    float*     mbuf  = (float*)    nxt((size_t)NN * 4 * 4);
    float*     dnm   = (float*)    nxt((size_t)NN * 4 * 4);
    _Float16*  pack  = (_Float16*) nxt((size_t)16 * 32 * 32 * 16 * 2);
    float*     sums  = (float*)    nxt((size_t)NG * HIDD * 4);
    float*     cnts  = (float*)    nxt((size_t)NG * 4);

    auto cdiv = [](long a, long b) { return (int)((a + b - 1) / b); };
    const int mtgGroups = (MTG + 7) / 8; // 49

    auto gemm = [&](const _Float16* A, int lda, int Kin, const float* Wm, const float* bi,
                    float* C, int Nout) {
        int Ktiles = (Kin + 31) / 32;
        int ntgCount = Nout / 32;
        int elems = Ktiles * (Nout / 16) * 512;
        k_prepack<<<cdiv(elems, 256), 256, 0, stream>>>(Wm, pack, Kin, Nout, Ktiles, elems);
        k_gemm_wmma<<<ntgCount * mtgGroups, 256, 0, stream>>>(A, lda, pack, Ktiles, Nout,
                                                              mtgGroups, bi, C);
    };

    auto attn = [&](int H, int S) {
        k_fill_f32<<<cdiv((long)NN * H, 256), 256, 0, stream>>>(mbuf, -INFINITY, NN * H);
        hipMemsetAsync(dnm, 0, (size_t)NN * H * 4, stream);
        k_edge_score<<<cdiv((long)NE * H, 8), 256, 0, stream>>>(qb, kb, ei, score, mbuf, H, S);
        k_exp_sum<<<cdiv((long)NE * H, 256), 256, 0, stream>>>(score, mbuf, dnm, ei, H);
        k_aggregate<<<cdiv((long)NE * H, 8), 256, 0, stream>>>(score, dnm, vb, ei, ob, H, S);
    };

    // ---- input features ----
    hipMemsetAsync(deg, 0, (size_t)NN * 4, stream);
    k_degree<<<cdiv(NE, 256), 256, 0, stream>>>(ei, deg);
    k_build_h0<<<cdiv((long)NNP * 160, 256), 256, 0, stream>>>(x, emb, deg, hf, 160);

    // ---- conv1: K=139 (pad 160) -> 512, H=4 ----
    gemm(hf, 160, 139, W[0], B[0], qb, 512);
    gemm(hf, 160, 139, W[1], B[1], kb, 512);
    gemm(hf, 160, 139, W[2], B[2], vb, 512);
    gemm(hf, 160, 139, W[3], B[3], ob, 512);   // skip into out
    attn(4, 512);
    k_elu_cvt<<<cdiv((long)NNP * 512, 256), 256, 0, stream>>>(ob, hf, NNP * 512);

    // ---- conv2: 512 -> 512, H=4 ----
    gemm(hf, 512, 512, W[4], B[4], qb, 512);
    gemm(hf, 512, 512, W[5], B[5], kb, 512);
    gemm(hf, 512, 512, W[6], B[6], vb, 512);
    gemm(hf, 512, 512, W[7], B[7], ob, 512);
    attn(4, 512);
    k_elu_cvt<<<cdiv((long)NNP * 512, 256), 256, 0, stream>>>(ob, hf, NNP * 512);

    // ---- conv3: 512 -> 128, H=1 (no ELU) ----
    gemm(hf, 512, 512, W[8],  B[8],  qb, 128);
    gemm(hf, 512, 512, W[9],  B[9],  kb, 128);
    gemm(hf, 512, 512, W[10], B[10], vb, 128);
    gemm(hf, 512, 512, W[11], B[11], ob, 128);
    attn(1, 128);

    // ---- mean pool + MLP head ----
    hipMemsetAsync(sums, 0, (size_t)NG * HIDD * 4, stream);
    hipMemsetAsync(cnts, 0, (size_t)NG * 4, stream);
    k_pool<<<cdiv((long)NN * HIDD, 256), 256, 0, stream>>>(ob, batch, sums, cnts);
    k_head<<<cdiv(NG, 8), 256, 0, stream>>>(sums, cnts, fc1w, fc1b, fc2w, fc2b, out);
}
